// MultiHeadAttentionLayer_65146063946020
// MI455X (gfx1250) — matrix operand order repaired and checked by reading, not compiled, not run on hardware
//
#include <hip/hip_runtime.h>

// ---------------------------------------------------------------------------
// MHA for MI455X (gfx1250): bf16 WMMA (v_wmma_f32_16x16x32_bf16) everywhere.
// B=2, S=2048, D=1024, H=16, DK=64.
// Round 3: f32->bf16 via v_cvt_pk_bf16_f32 (builtin if declared, else inline
// asm) instead of the 3-op manual RNE sequence.  Pipeline structure unchanged:
// double-buffered LDS, 1 barrier per K-step, register-staged global prefetch.
// ---------------------------------------------------------------------------

typedef __attribute__((ext_vector_type(16))) __bf16 v16bf;
typedef __attribute__((ext_vector_type(2)))  __bf16 v2bf;
typedef __attribute__((ext_vector_type(8)))  float  v8f;
typedef unsigned short u16;
typedef unsigned int   u32;

constexpr int Bb = 2, Ss = 2048, Dd = 1024, Hh = 16, DKk = 64;
constexpr int Mrows = Bb * Ss;  // 4096

union BFrag  { v16bf v; u16 u[16]; uint4 q[2]; };
union Pack16 { uint4 q[2]; u16 u[16]; };
union Pack8  { uint4 q;    u16 u[8];  };

#if defined(__has_builtin)
#if __has_builtin(__builtin_amdgcn_cvt_pk_bf16_f32)
#define HAVE_CVT_PK_BF16 1
#endif
#endif

// pack two floats into two bf16 (RNE) in one dword
__device__ inline u32 pack2bf(float a, float b) {
#ifdef HAVE_CVT_PK_BF16
  union { v2bf v; u32 u; } r;
  r.v = __builtin_amdgcn_cvt_pk_bf16_f32(a, b);
  return r.u;
#else
  u32 r;
  asm("v_cvt_pk_bf16_f32 %0, %1, %2" : "=v"(r) : "v"(a), "v"(b));
  return r;
#endif
}

__device__ inline u16 f2bf(float f) { return (u16)pack2bf(f, f); }

// convert float4 -> 4 consecutive bf16 at dst (dst 4-byte aligned)
__device__ inline void cvt4(const float4& f, u16* dst) {
  *reinterpret_cast<u32*>(dst)     = pack2bf(f.x, f.y);
  *reinterpret_cast<u32*>(dst + 2) = pack2bf(f.z, f.w);
}

__device__ inline v8f zero8() {
  v8f z = {0.f, 0.f, 0.f, 0.f, 0.f, 0.f, 0.f, 0.f};
  return z;
}

__device__ inline v8f wmma_bf16(const BFrag& a, const BFrag& b, v8f c) {
  return __builtin_amdgcn_wmma_f32_16x16x32_bf16(
      /*neg_a=*/false, a.v, /*neg_b=*/false, b.v,
      /*c_mod=*/(short)0, c, /*reuse_a=*/false, /*reuse_b=*/false);
}

// ---------------------------------------------------------------------------
// Kernel 1: Q/K/V projections.  C[m,n] = sum_k A[m,k] * W[n,k]  (y = x @ W.T)
// blockIdx.z selects q/k/v + Wq/Wk/Wv.  Output bf16.
// WG tile 128x128, 8 waves (2x4), wave tile 64x32 (4x2 accumulators), BK=32.
// Double-buffered LDS; f32->bf16 conversion at LDS-store time.
// ---------------------------------------------------------------------------
__global__ __launch_bounds__(256)
void proj_gemm_kernel(const float* __restrict__ qin, const float* __restrict__ kin,
                      const float* __restrict__ vin,
                      const float* __restrict__ Wq, const float* __restrict__ Wk,
                      const float* __restrict__ Wv,
                      u16* __restrict__ Qp, u16* __restrict__ Kp, u16* __restrict__ Vp) {
  __shared__ u16 lA[2][128][40];  // 128 rows x 32 k, pad to 40
  __shared__ u16 lB[2][128][40];

  const int z = blockIdx.z;
  const float* A = (z == 0) ? qin : (z == 1) ? kin : vin;
  const float* W = (z == 0) ? Wq  : (z == 1) ? Wk  : Wv;
  u16* O        = (z == 0) ? Qp  : (z == 1) ? Kp  : Vp;

  const int tid  = threadIdx.x;
  const int wave = tid >> 5, lane = tid & 31;
  const int laneh = lane & 15, half = lane >> 4;
  const int wm = wave >> 2, wn = wave & 3;          // wave grid 2 x 4
  const int m0 = blockIdx.y * 128, n0 = blockIdx.x * 128;
  const int srow = tid >> 1, shalf = tid & 1;       // staging: 128 rows x 2 halves

  const float* aRow = &A[(size_t)(m0 + srow) * Dd + shalf * 16];
  const float* bRow = &W[(size_t)(n0 + srow) * Dd + shalf * 16];

  v8f acc[4][2];
#pragma unroll
  for (int i = 0; i < 4; ++i)
#pragma unroll
    for (int j = 0; j < 2; ++j) acc[i][j] = zero8();

  constexpr int NT = Dd / 32;
  float4 ra[4], rb[4];
  {
    const float4* sa = reinterpret_cast<const float4*>(aRow);
    const float4* sb = reinterpret_cast<const float4*>(bRow);
#pragma unroll
    for (int i = 0; i < 4; ++i) { ra[i] = sa[i]; rb[i] = sb[i]; }
  }

  for (int kt = 0; kt < NT; ++kt) {
    const int buf = kt & 1;
    // ---- convert + store staged regs to LDS[buf] ----
    {
      Pack16 pa, pb;
#pragma unroll
      for (int i = 0; i < 4; ++i) { cvt4(ra[i], &pa.u[4 * i]); cvt4(rb[i], &pb.u[4 * i]); }
      *reinterpret_cast<uint4*>(&lA[buf][srow][shalf * 16])     = pa.q[0];
      *reinterpret_cast<uint4*>(&lA[buf][srow][shalf * 16 + 8]) = pa.q[1];
      *reinterpret_cast<uint4*>(&lB[buf][srow][shalf * 16])     = pb.q[0];
      *reinterpret_cast<uint4*>(&lB[buf][srow][shalf * 16 + 8]) = pb.q[1];
    }
    __syncthreads();

    // ---- prefetch next tile into registers (overlaps compute) ----
    if (kt + 1 < NT) {
      const float4* sa = reinterpret_cast<const float4*>(aRow + (kt + 1) * 32);
      const float4* sb = reinterpret_cast<const float4*>(bRow + (kt + 1) * 32);
#pragma unroll
      for (int i = 0; i < 4; ++i) { ra[i] = sa[i]; rb[i] = sb[i]; }
    }

    // ---- fragments + 8 WMMAs ----
    BFrag af[4], bfr[2];
    const int abase = half ? 8 : 0;   // A lanes 16-31 hold K {8..15, 24..31}
#pragma unroll
    for (int i = 0; i < 4; ++i) {
      const int r = wm * 64 + i * 16 + laneh;
      af[i].q[0] = *reinterpret_cast<const uint4*>(&lA[buf][r][abase]);
      af[i].q[1] = *reinterpret_cast<const uint4*>(&lA[buf][r][abase + 16]);
    }
    const int bbase = half ? 8 : 0;   // B lanes 16-31 hold K {8..15, 24..31}
#pragma unroll
    for (int j = 0; j < 2; ++j) {
      const int r = wn * 32 + j * 16 + laneh;
      bfr[j].q[0] = *reinterpret_cast<const uint4*>(&lB[buf][r][bbase]);
      bfr[j].q[1] = *reinterpret_cast<const uint4*>(&lB[buf][r][bbase + 16]);
    }
#pragma unroll
    for (int i = 0; i < 4; ++i)
#pragma unroll
      for (int j = 0; j < 2; ++j)
        acc[i][j] = wmma_bf16(af[i], bfr[j], acc[i][j]);
  }

  // ---- store bf16 (C layout: vgpr r -> row r + 8*half, col = laneh) ----
#pragma unroll
  for (int i = 0; i < 4; ++i)
#pragma unroll
    for (int j = 0; j < 2; ++j)
#pragma unroll
      for (int r = 0; r < 8; ++r) {
        const int row = m0 + wm * 64 + i * 16 + r + half * 8;
        const int col = n0 + wn * 32 + j * 16 + laneh;
        O[(size_t)row * Dd + col] = f2bf(acc[i][j][r]);
      }
}

// ---------------------------------------------------------------------------
// Kernel 2: flash attention per (b,h).  WG = 128 q-rows (16 per wave),
// loop over 32-row K/V blocks, double-buffered in LDS.  V staged transposed.
// ---------------------------------------------------------------------------
__global__ __launch_bounds__(256)
void attn_kernel(const u16* __restrict__ Qp, const u16* __restrict__ Kp,
                 const u16* __restrict__ Vp, const int* __restrict__ mask,
                 u16* __restrict__ ctx) {
  __shared__ u16 lK[2][32][72];    // K block: 32 rows x 64 d (padded)
  __shared__ u16 lV[2][64][40];    // V block transposed: 64 d x 32 k (padded)
  __shared__ u16 lP[8][16][40];    // per-wave P tile: 16 x 32 (padded)

  const int tid  = threadIdx.x;
  const int wave = tid >> 5, lane = tid & 31;
  const int laneh = lane & 15, half = lane >> 4;
  const int qb = blockIdx.x;
  const int bh = blockIdx.y;
  const int b = bh / Hh, h = bh % Hh;
  const int hcol = h * DKk;

  // persistent Q fragments (16 rows x 64 d = two K-chunks of 32)
  const int qrow = qb * 128 + wave * 16 + laneh;
  const u16* qptr = &Qp[(size_t)(b * Ss + qrow) * Dd + hcol];
  BFrag qf[2];
#pragma unroll
  for (int c = 0; c < 2; ++c) {
    const int base = c * 32 + (half ? 8 : 0);
    qf[c].q[0] = *reinterpret_cast<const uint4*>(qptr + base);
    qf[c].q[1] = *reinterpret_cast<const uint4*>(qptr + base + 16);
  }

  float mrow[8], lrow[8];
  v8f cacc[4];
#pragma unroll
  for (int r = 0; r < 8; ++r) { mrow[r] = -1e30f; lrow[r] = 0.f; }
#pragma unroll
  for (int t = 0; t < 4; ++t) cacc[t] = zero8();

  const int srow = tid >> 3;  // 0..31  (k-row within block)
  const int sseg = tid & 7;   // 0..7   (8-element d segment)
  const u16* kRow = &Kp[(size_t)(b * Ss + srow) * Dd + hcol + sseg * 8];
  const u16* vRow = &Vp[(size_t)(b * Ss + srow) * Dd + hcol + sseg * 8];

  constexpr int NB = Ss / 32;
  Pack8 rk, rv;
  rk.q = *reinterpret_cast<const uint4*>(kRow);
  rv.q = *reinterpret_cast<const uint4*>(vRow);

  for (int kb = 0; kb < NB; ++kb) {
    const int buf = kb & 1;
    const int kr0 = kb * 32;

    // ---- store staged K (row-major) and V (transposed) into LDS[buf] ----
    *reinterpret_cast<uint4*>(&lK[buf][srow][sseg * 8]) = rk.q;
#pragma unroll
    for (int e = 0; e < 8; ++e) lV[buf][sseg * 8 + e][srow] = rv.u[e];
    __syncthreads();

    // ---- prefetch next K/V block into registers ----
    if (kb + 1 < NB) {
      rk.q = *reinterpret_cast<const uint4*>(kRow + (size_t)(kr0 + 32) * Dd);
      rv.q = *reinterpret_cast<const uint4*>(vRow + (size_t)(kr0 + 32) * Dd);
    }

    // ---- scores: 16 q-rows x 32 k-cols, contraction over d=64 ----
    v8f s0 = zero8(), s1 = zero8();
#pragma unroll
    for (int c = 0; c < 2; ++c) {
      const int dbase = c * 32 + (half ? 8 : 0);
      BFrag kf0, kf1;
      kf0.q[0] = *reinterpret_cast<const uint4*>(&lK[buf][laneh][dbase]);
      kf0.q[1] = *reinterpret_cast<const uint4*>(&lK[buf][laneh][dbase + 16]);
      kf1.q[0] = *reinterpret_cast<const uint4*>(&lK[buf][16 + laneh][dbase]);
      kf1.q[1] = *reinterpret_cast<const uint4*>(&lK[buf][16 + laneh][dbase + 16]);
      s0 = wmma_bf16(qf[c], kf0, s0);
      s1 = wmma_bf16(qf[c], kf1, s1);
    }

    const float mb0 = mask[b * Ss + kr0 + laneh]      ? 0.f : -1e9f;
    const float mb1 = mask[b * Ss + kr0 + 16 + laneh] ? 0.f : -1e9f;

    float p0[8], p1[8], mx[8];
#pragma unroll
    for (int r = 0; r < 8; ++r) {
      p0[r] = s0[r] * 0.125f + mb0;   // 1/sqrt(64), then mask
      p1[r] = s1[r] * 0.125f + mb1;
      mx[r] = fmaxf(p0[r], p1[r]);
    }
#pragma unroll
    for (int off = 1; off < 16; off <<= 1)
#pragma unroll
      for (int r = 0; r < 8; ++r)
        mx[r] = fmaxf(mx[r], __shfl_xor(mx[r], off, 32));

    float alpha[8], rs[8];
#pragma unroll
    for (int r = 0; r < 8; ++r) {
      const float mnew = fmaxf(mrow[r], mx[r]);
      alpha[r] = __expf(mrow[r] - mnew);
      mrow[r] = mnew;
      p0[r] = __expf(p0[r] - mnew);
      p1[r] = __expf(p1[r] - mnew);
      rs[r] = p0[r] + p1[r];
    }
#pragma unroll
    for (int off = 1; off < 16; off <<= 1)
#pragma unroll
      for (int r = 0; r < 8; ++r)
        rs[r] += __shfl_xor(rs[r], off, 32);
#pragma unroll
    for (int r = 0; r < 8; ++r) {
      lrow[r] = lrow[r] * alpha[r] + rs[r];
#pragma unroll
      for (int t = 0; t < 4; ++t) cacc[t][r] *= alpha[r];
    }

    // ---- P: C-layout -> LDS -> A-layout ----
#pragma unroll
    for (int r = 0; r < 8; ++r) {
      lP[wave][r + half * 8][laneh]      = f2bf(p0[r]);
      lP[wave][r + half * 8][16 + laneh] = f2bf(p1[r]);
    }
    __syncthreads();

    BFrag pf;
    {
      const int base = half ? 8 : 0;
      pf.q[0] = *reinterpret_cast<const uint4*>(&lP[wave][laneh][base]);
      pf.q[1] = *reinterpret_cast<const uint4*>(&lP[wave][laneh][base + 16]);
    }
    // ---- ctx += P @ V  (4 d-subtiles of 16) ----
    const int kbase = half ? 8 : 0;
#pragma unroll
    for (int t = 0; t < 4; ++t) {
      BFrag vf;
      vf.q[0] = *reinterpret_cast<const uint4*>(&lV[buf][t * 16 + laneh][kbase]);
      vf.q[1] = *reinterpret_cast<const uint4*>(&lV[buf][t * 16 + laneh][kbase + 16]);
      cacc[t] = wmma_bf16(pf, vf, cacc[t]);
    }
  }

  // ---- normalize and store ctx (bf16, [B,S,D] with col = h*64 + d) ----
#pragma unroll
  for (int t = 0; t < 4; ++t)
#pragma unroll
    for (int r = 0; r < 8; ++r) {
      const float val = cacc[t][r] / lrow[r];
      const int row = qb * 128 + wave * 16 + r + half * 8;
      const int col = hcol + t * 16 + laneh;
      ctx[(size_t)(b * Ss + row) * Dd + col] = f2bf(val);
    }
}

// ---------------------------------------------------------------------------
// Kernel 3: output projection.  out[m,n] = sum_k ctx[m,k] * Wo[n,k], f32 out.
// Same double-buffered skeleton; A is already bf16.
// ---------------------------------------------------------------------------
__global__ __launch_bounds__(256)
void out_gemm_kernel(const u16* __restrict__ Actx, const float* __restrict__ Wo,
                     float* __restrict__ out) {
  __shared__ u16 lA[2][128][40];
  __shared__ u16 lB[2][128][40];

  const int tid  = threadIdx.x;
  const int wave = tid >> 5, lane = tid & 31;
  const int laneh = lane & 15, half = lane >> 4;
  const int wm = wave >> 2, wn = wave & 3;
  const int m0 = blockIdx.y * 128, n0 = blockIdx.x * 128;
  const int srow = tid >> 1, shalf = tid & 1;

  const u16*   aRow = &Actx[(size_t)(m0 + srow) * Dd + shalf * 16];
  const float* bRow = &Wo[(size_t)(n0 + srow) * Dd + shalf * 16];

  v8f acc[4][2];
#pragma unroll
  for (int i = 0; i < 4; ++i)
#pragma unroll
    for (int j = 0; j < 2; ++j) acc[i][j] = zero8();

  constexpr int NT = Dd / 32;
  uint4  ra[2];
  float4 rb[4];
  {
    const uint4*  sa = reinterpret_cast<const uint4*>(aRow);
    const float4* sb = reinterpret_cast<const float4*>(bRow);
    ra[0] = sa[0]; ra[1] = sa[1];
#pragma unroll
    for (int i = 0; i < 4; ++i) rb[i] = sb[i];
  }

  for (int kt = 0; kt < NT; ++kt) {
    const int buf = kt & 1;
    {
      Pack16 pb;
#pragma unroll
      for (int i = 0; i < 4; ++i) cvt4(rb[i], &pb.u[4 * i]);
      *reinterpret_cast<uint4*>(&lA[buf][srow][shalf * 16])     = ra[0];
      *reinterpret_cast<uint4*>(&lA[buf][srow][shalf * 16 + 8]) = ra[1];
      *reinterpret_cast<uint4*>(&lB[buf][srow][shalf * 16])     = pb.q[0];
      *reinterpret_cast<uint4*>(&lB[buf][srow][shalf * 16 + 8]) = pb.q[1];
    }
    __syncthreads();

    if (kt + 1 < NT) {
      const uint4*  sa = reinterpret_cast<const uint4*>(aRow + (kt + 1) * 32);
      const float4* sb = reinterpret_cast<const float4*>(bRow + (kt + 1) * 32);
      ra[0] = sa[0]; ra[1] = sa[1];
#pragma unroll
      for (int i = 0; i < 4; ++i) rb[i] = sb[i];
    }

    BFrag af[4], bfr[2];
    const int abase = half ? 8 : 0;
#pragma unroll
    for (int i = 0; i < 4; ++i) {
      const int r = wm * 64 + i * 16 + laneh;
      af[i].q[0] = *reinterpret_cast<const uint4*>(&lA[buf][r][abase]);
      af[i].q[1] = *reinterpret_cast<const uint4*>(&lA[buf][r][abase + 16]);
    }
    const int bbase = half ? 8 : 0;
#pragma unroll
    for (int j = 0; j < 2; ++j) {
      const int r = wn * 32 + j * 16 + laneh;
      bfr[j].q[0] = *reinterpret_cast<const uint4*>(&lB[buf][r][bbase]);
      bfr[j].q[1] = *reinterpret_cast<const uint4*>(&lB[buf][r][bbase + 16]);
    }
#pragma unroll
    for (int i = 0; i < 4; ++i)
#pragma unroll
      for (int j = 0; j < 2; ++j)
        acc[i][j] = wmma_bf16(af[i], bfr[j], acc[i][j]);
  }

#pragma unroll
  for (int i = 0; i < 4; ++i)
#pragma unroll
    for (int j = 0; j < 2; ++j)
#pragma unroll
      for (int r = 0; r < 8; ++r) {
        const int row = m0 + wm * 64 + i * 16 + r + half * 8;
        const int col = n0 + wn * 32 + j * 16 + laneh;
        out[(size_t)row * Dd + col] = acc[i][j][r];
      }
}

// ---------------------------------------------------------------------------
// Launch: proj (Q/K/V fused in grid.z) -> attention -> output projection.
// Workspace: 4 bf16 buffers of B*S*D (Qp, Kp, Vp, ctx) = 32 MB.
// ---------------------------------------------------------------------------
extern "C" void kernel_launch(void* const* d_in, const int* in_sizes, int n_in,
                              void* d_out, int out_size, void* d_ws, size_t ws_size,
                              hipStream_t stream) {
  const float* q    = (const float*)d_in[0];
  const float* k    = (const float*)d_in[1];
  const float* v    = (const float*)d_in[2];
  const int*   mask = (const int*)  d_in[3];
  const float* Wq   = (const float*)d_in[4];
  const float* Wk   = (const float*)d_in[5];
  const float* Wv   = (const float*)d_in[6];
  const float* Wo   = (const float*)d_in[7];
  float* out = (float*)d_out;

  u16* ws = (u16*)d_ws;
  const size_t elems = (size_t)Bb * Ss * Dd;  // 4Mi elements
  u16* Qp  = ws;
  u16* Kp  = ws + elems;
  u16* Vp  = ws + 2 * elems;
  u16* Ctx = ws + 3 * elems;

  dim3 blk(256);
  proj_gemm_kernel<<<dim3(Dd / 128, Mrows / 128, 3), blk, 0, stream>>>(
      q, k, v, Wq, Wk, Wv, Qp, Kp, Vp);
  attn_kernel<<<dim3(Ss / 128, Bb * Hh), blk, 0, stream>>>(Qp, Kp, Vp, mask, Ctx);
  out_gemm_kernel<<<dim3(Dd / 128, Mrows / 128), blk, 0, stream>>>(Ctx, Wo, out);
}
